// Attention_7988639170930
// MI455X (gfx1250) — compile-verified, matrix operand order checked by reading, not run on hardware
//
#include <hip/hip_runtime.h>
#include <hip/hip_bf16.h>

// ---------------------------------------------------------------------------
// MI455X (gfx1250) attention block: bf16 WMMA GEMMs + fused flash attention.
// wave32; v_wmma_f32_16x16x32_bf16 for all matrix math; async global->LDS
// (ASYNCcnt) staging with double-buffered LDS in the GEMM.
// ---------------------------------------------------------------------------

typedef __attribute__((ext_vector_type(16))) __bf16 v16bf;
typedef __attribute__((ext_vector_type(8)))  __bf16 v8bf;
typedef __attribute__((ext_vector_type(8)))  float  v8f;
typedef __attribute__((ext_vector_type(4)))  int    v4i;

union AFrag { v16bf v; v8bf h[2]; };

__device__ __forceinline__ v8f wmma_bf16(const AFrag& a, const AFrag& b, v8f c) {
  // 8 args: (neg_a, A, neg_b, B, c_mod, C, reuse_a, reuse_b)
  return __builtin_amdgcn_wmma_f32_16x16x32_bf16(false, a.v, false, b.v,
                                                 (short)0, c, false, false);
}

// ---- async global -> LDS, 16B per lane (GLOBAL_LOAD_ASYNC_TO_LDS_B128) ----
// Probe-confirmed param 0 type: non-const v4i in AS(1); param 1: v4i in AS(3).
__device__ __forceinline__ void async_copy_b128(void* lds, const void* g) {
#if __has_builtin(__builtin_amdgcn_global_load_async_to_lds_b128)
  __builtin_amdgcn_global_load_async_to_lds_b128(
      (__attribute__((address_space(1))) v4i*)g,
      (__attribute__((address_space(3))) v4i*)lds, 0, 0);
#else
  asm volatile("global_load_async_to_lds_b128 %0, %1, off"
               :: "v"((unsigned)(size_t)lds),
                  "v"((unsigned long long)(size_t)g)
               : "memory");
#endif
}

__device__ __forceinline__ void wait_async0() {
#if __has_builtin(__builtin_amdgcn_s_wait_asynccnt)
  __builtin_amdgcn_s_wait_asynccnt(0);
#else
  asm volatile("s_wait_asynccnt 0x0" ::: "memory");
#endif
}

// ---------------------------------------------------------------------------
// fp32 -> bf16 convert (grid-stride)
// ---------------------------------------------------------------------------
__global__ void f32_to_bf16_kernel(const float* __restrict__ in,
                                   __bf16* __restrict__ out, long n) {
  long i = (long)blockIdx.x * blockDim.x + threadIdx.x;
  long stride = (long)gridDim.x * blockDim.x;
  for (; i < n; i += stride) out[i] = (__bf16)in[i];
}

// ---------------------------------------------------------------------------
// RoPE in place on bf16 [B*L][H*128]; cos/sin are [L][128] (concat(freqs,freqs))
// ---------------------------------------------------------------------------
__global__ void rope_kernel(__bf16* __restrict__ X, const float* __restrict__ cosT,
                            const float* __restrict__ sinT, int L, int H, long total) {
  long idx = (long)blockIdx.x * blockDim.x + threadIdx.x;
  if (idx >= total) return;
  int d = (int)(idx & 63);
  int h = (int)((idx >> 6) % H);
  long row = idx / ((long)64 * H);
  int l = (int)(row % L);
  float c0 = cosT[l * 128 + d],      s0 = sinT[l * 128 + d];
  float c1 = cosT[l * 128 + d + 64], s1 = sinT[l * 128 + d + 64];
  __bf16* p = X + row * ((long)H * 128) + (long)h * 128;
  float x0 = (float)p[d], x1 = (float)p[d + 64];
  p[d]      = (__bf16)(x0 * c0 - x1 * s0);
  p[d + 64] = (__bf16)(x1 * c1 + x0 * s1);
}

// ---------------------------------------------------------------------------
// bf16 GEMM: C[M,N] = A[M,K] @ B[K,N]; block tile 128x128, K-step 32,
// double-buffered LDS, async A staging. 8 waves in 4x2; wave -> 32x64
// (2x4 WMMA tiles, 8 v_wmma per K-step vs 12 ds_load_b128 fragment loads).
// ---------------------------------------------------------------------------
template <bool OUT_F32>
__global__ __launch_bounds__(256) void gemm_bf16_wmma(
    const __bf16* __restrict__ A, const __bf16* __restrict__ Bm,
    float* __restrict__ Cf, __bf16* __restrict__ Cb,
    int M, int N, int K) {
  constexpr int BM = 128, BN = 128, BK = 32;
  __shared__ __align__(16) __bf16 sA[2][BM][BK + 8];   // [m][k], 80B row stride
  __shared__ __align__(16) __bf16 sB[2][BN][BK + 8];   // [n][k] (transposed)

  const int tid = threadIdx.x;
  const int lane = tid & 31, wave = tid >> 5;
  const int wr = wave >> 1, wc = wave & 1;             // 4x2 wave grid
  const int m0 = blockIdx.y * BM, n0 = blockIdx.x * BN;
  const int r  = lane & 15;
  const int hi = lane >> 4;
  const int kh = hi * 8;                               // element offset in half

  // staging thread mapping
  const int arow = tid >> 1, aseg = (tid & 1) * 16;    // A: 2 thr/row, 16 elems
  const int bkk = tid >> 3, bnc = (tid & 7) * 16;      // B: 8 thr/k-row, 16 cols

  v8f acc[2][4] = {};

  const int nk = K / BK;

  // ---- prologue: stage tile 0 into buffer 0 ----
  {
    const __bf16* ga = A + (size_t)(m0 + arow) * K + aseg;
    async_copy_b128(&sA[0][arow][aseg],     ga);
    async_copy_b128(&sA[0][arow][aseg + 8], ga + 8);
    const __bf16* gb = Bm + (size_t)bkk * N + n0 + bnc;
    v8bf b0 = *(const v8bf*)gb;
    v8bf b1 = *(const v8bf*)(gb + 8);
#pragma unroll
    for (int i = 0; i < 8; i++) { sB[0][bnc + i][bkk] = b0[i]; sB[0][bnc + 8 + i][bkk] = b1[i]; }
  }
  wait_async0();
  __syncthreads();

  for (int kt = 0; kt < nk; ++kt) {
    const int cur = kt & 1, nxt = cur ^ 1;
    v8bf b0, b1;
    if (kt + 1 < nk) {
      // issue next tile's global traffic before compute: A async, B to VGPRs
      const int k0 = (kt + 1) * BK;
      const __bf16* ga = A + (size_t)(m0 + arow) * K + k0 + aseg;
      async_copy_b128(&sA[nxt][arow][aseg],     ga);
      async_copy_b128(&sA[nxt][arow][aseg + 8], ga + 8);
      const __bf16* gb = Bm + (size_t)(k0 + bkk) * N + n0 + bnc;
      b0 = *(const v8bf*)gb;
      b1 = *(const v8bf*)(gb + 8);
    }

    // ---- compute current tile ----
    AFrag a[2], bf[4];
#pragma unroll
    for (int mt = 0; mt < 2; mt++) {
      int mr = wr * 32 + mt * 16 + r;
      a[mt].h[0] = *(const v8bf*)&sA[cur][mr][kh];
      a[mt].h[1] = *(const v8bf*)&sA[cur][mr][16 + kh];
    }
#pragma unroll
    for (int nt = 0; nt < 4; nt++) {
      int nc = wc * 64 + nt * 16 + r;
      bf[nt].h[0] = *(const v8bf*)&sB[cur][nc][kh];
      bf[nt].h[1] = *(const v8bf*)&sB[cur][nc][16 + kh];
    }
#pragma unroll
    for (int mt = 0; mt < 2; mt++)
#pragma unroll
      for (int nt = 0; nt < 4; nt++)
        acc[mt][nt] = wmma_bf16(a[mt], bf[nt], acc[mt][nt]);

    if (kt + 1 < nk) {
      // scatter-transpose next B tile, then make next buffer visible
#pragma unroll
      for (int i = 0; i < 8; i++) { sB[nxt][bnc + i][bkk] = b0[i]; sB[nxt][bnc + 8 + i][bkk] = b1[i]; }
      wait_async0();
      __syncthreads();
    }
  }

  // ---- epilogue: C layout = VGPR i: rows (i | i+8), cols = lane&15 ----
#pragma unroll
  for (int mt = 0; mt < 2; mt++)
#pragma unroll
    for (int nt = 0; nt < 4; nt++)
#pragma unroll
      for (int i = 0; i < 8; i++) {
        int row = m0 + wr * 32 + mt * 16 + i + 8 * hi;
        int col = n0 + wc * 64 + nt * 16 + r;
        float v = acc[mt][nt][i];
        if (OUT_F32) Cf[(size_t)row * N + col] = v;
        else         Cb[(size_t)row * N + col] = (__bf16)v;
      }
}

// ---------------------------------------------------------------------------
// Flash attention (causal), GQA 16 heads / 4 kv heads, Dh=128.
// Block = 8 waves, 128 q rows; wave owns 16 q rows (Q held as 4 A-frags in
// registers). kv tiles of 32; online softmax; P round-trips via per-wave LDS.
// K tile staged with async global->LDS; V tile scatter-transposed.
// ---------------------------------------------------------------------------
__global__ __launch_bounds__(256) void flash_attn_kernel(
    const __bf16* __restrict__ Q, const __bf16* __restrict__ Kv,
    const __bf16* __restrict__ Vv, __bf16* __restrict__ O, int L) {
  constexpr int Dh = 128, H = 16, KVH = 4;
  const int ldq = H * Dh;     // 2048
  const int ldkv = KVH * Dh;  // 512

  __shared__ __align__(16) __bf16 sK[32][Dh + 8];        // [kv][d]
  __shared__ __align__(16) __bf16 sV[Dh][32 + 8];        // [d][kv] (transposed)
  __shared__ __align__(16) __bf16 sP[8][16][32 + 8];     // per-wave P scratch

  const int tid = threadIdx.x, lane = tid & 31, wave = tid >> 5;
  const int q0 = blockIdx.x * 128;
  const int h = blockIdx.y, b = blockIdx.z;
  const int kvh = h / (H / KVH);
  const int r = lane & 15, hi = lane >> 4, kh = hi * 8;
  const float scale = 0.08838834764831845f;              // 1/sqrt(128)
  const float NEG_INF = -__builtin_inff();

  // Q fragments: 16 rows x 128 d = 4 A-frags of 16x32
  AFrag qa[4];
  {
    const __bf16* qp = Q + (size_t)(b * L + q0 + wave * 16 + r) * ldq + h * Dh;
#pragma unroll
    for (int kk = 0; kk < 4; kk++) {
      qa[kk].h[0] = *(const v8bf*)(qp + kk * 32 + kh);
      qa[kk].h[1] = *(const v8bf*)(qp + kk * 32 + 16 + kh);
    }
  }

  float mrow[8], lrow[8];
  v8f o[8] = {};
#pragma unroll
  for (int i = 0; i < 8; i++) { mrow[i] = NEG_INF; lrow[i] = 0.f; }

  const int njt = (q0 + 128) / 32;                       // causal bound
  for (int j = 0; j < njt; j++) {
    const int jb = j * 32;
    __syncthreads();                                     // protect prior sK/sV reads
    // ---- stage K (async, row major) and V (transposed) ----
    {
      int kv = tid >> 3, seg = (tid & 7) * 16;
      const __bf16* gk = Kv + (size_t)(b * L + jb + kv) * ldkv + kvh * Dh + seg;
      async_copy_b128(&sK[kv][seg],     gk);
      async_copy_b128(&sK[kv][seg + 8], gk + 8);
      const __bf16* gv = Vv + (size_t)(b * L + jb + kv) * ldkv + kvh * Dh + seg;
      v8bf v0 = *(const v8bf*)gv;
      v8bf v1 = *(const v8bf*)(gv + 8);
#pragma unroll
      for (int i = 0; i < 8; i++) { sV[seg + i][kv] = v0[i]; sV[seg + 8 + i][kv] = v1[i]; }
    }
    wait_async0();
    __syncthreads();

    // ---- S = Q @ K^T (two 16x16 tiles over kv 0..31) ----
    v8f s0 = {}, s1 = {};
#pragma unroll
    for (int kk = 0; kk < 4; kk++) {
      AFrag b0, b1;
      b0.h[0] = *(const v8bf*)&sK[r][kk * 32 + kh];
      b0.h[1] = *(const v8bf*)&sK[r][kk * 32 + 16 + kh];
      b1.h[0] = *(const v8bf*)&sK[16 + r][kk * 32 + kh];
      b1.h[1] = *(const v8bf*)&sK[16 + r][kk * 32 + 16 + kh];
      s0 = wmma_bf16(qa[kk], b0, s0);
      s1 = wmma_bf16(qa[kk], b1, s1);
    }

    // ---- causal mask + online softmax (rows live across 16-lane groups) ----
#pragma unroll
    for (int i = 0; i < 8; i++) {
      int row = q0 + wave * 16 + i + 8 * hi;
      int c0 = jb + r, c1 = c0 + 16;
      float v0 = s0[i] * scale, v1 = s1[i] * scale;
      if (c0 > row) v0 = NEG_INF;
      if (c1 > row) v1 = NEG_INF;
      float t = fmaxf(v0, v1);
#pragma unroll
      for (int msk = 1; msk < 16; msk <<= 1) t = fmaxf(t, __shfl_xor(t, msk, 32));
      float mnew = fmaxf(mrow[i], t);
      float sc = __expf(mrow[i] - mnew);
      float p0 = __expf(v0 - mnew), p1 = __expf(v1 - mnew);
      float ps = p0 + p1;
#pragma unroll
      for (int msk = 1; msk < 16; msk <<= 1) ps += __shfl_xor(ps, msk, 32);
      lrow[i] = lrow[i] * sc + ps;
      mrow[i] = mnew;
#pragma unroll
      for (int dt = 0; dt < 8; dt++) o[dt][i] *= sc;
      sP[wave][i + 8 * hi][r]      = (__bf16)p0;   // C-layout -> LDS row-major
      sP[wave][i + 8 * hi][16 + r] = (__bf16)p1;
    }

    // ---- reload P in A-frag layout, O += P @ V ----
    AFrag pa;
    pa.h[0] = *(const v8bf*)&sP[wave][r][kh];
    pa.h[1] = *(const v8bf*)&sP[wave][r][16 + kh];
#pragma unroll
    for (int dt = 0; dt < 8; dt++) {
      AFrag bv;
      bv.h[0] = *(const v8bf*)&sV[dt * 16 + r][kh];
      bv.h[1] = *(const v8bf*)&sV[dt * 16 + r][16 + kh];
      o[dt] = wmma_bf16(pa, bv, o[dt]);
    }
  }

  // ---- normalize and store (bf16, [B*L][H*Dh]) ----
#pragma unroll
  for (int dt = 0; dt < 8; dt++)
#pragma unroll
    for (int i = 0; i < 8; i++) {
      int row = q0 + wave * 16 + i + 8 * hi;
      O[(size_t)(b * L + row) * ldq + h * Dh + dt * 16 + r] =
          (__bf16)(o[dt][i] / lrow[i]);
    }
}

// ---------------------------------------------------------------------------
// Orchestration
// ---------------------------------------------------------------------------
extern "C" void kernel_launch(void* const* d_in, const int* in_sizes, int n_in,
                              void* d_out, int out_size, void* d_ws, size_t ws_size,
                              hipStream_t stream) {
  const float* hs   = (const float*)d_in[0];
  const float* cosT = (const float*)d_in[1];
  const float* sinT = (const float*)d_in[2];
  const float* Wq   = (const float*)d_in[3];
  const float* Wk   = (const float*)d_in[4];
  const float* Wv   = (const float*)d_in[5];
  const float* Wo   = (const float*)d_in[6];
  float* out = (float*)d_out;

  const int B = 2, L = 2048, D = 2048, H = 16, KVH = 4, Dh = 128;
  const int M = B * L;        // 4096
  const int Nq = H * Dh;      // 2048
  const int Nkv = KVH * Dh;   // 512

  // Workspace layout (~48 MB). Attention output aliases Xb (X dead after QKV).
  char* ws = (char*)d_ws;
  __bf16* Xb = (__bf16*)ws;  ws += (size_t)M * D * 2;       // 16 MB
  __bf16* Wb = (__bf16*)ws;  ws += (size_t)D * Nq * 2;      //  8 MB (reused per weight)
  __bf16* Qb = (__bf16*)ws;  ws += (size_t)M * Nq * 2;      // 16 MB
  __bf16* Kb = (__bf16*)ws;  ws += (size_t)M * Nkv * 2;     //  4 MB
  __bf16* Vb = (__bf16*)ws;  ws += (size_t)M * Nkv * 2;     //  4 MB
  __bf16* Ab = Xb;                                          // alias

  dim3 blk(256);

  // 1) activations -> bf16
  f32_to_bf16_kernel<<<4096, blk, 0, stream>>>(hs, Xb, (long)M * D);

  // 2) Q = X @ Wq
  f32_to_bf16_kernel<<<4096, blk, 0, stream>>>(Wq, Wb, (long)D * Nq);
  gemm_bf16_wmma<false><<<dim3(Nq / 128, M / 128), blk, 0, stream>>>(
      Xb, Wb, nullptr, Qb, M, Nq, D);

  // 3) K = X @ Wk
  f32_to_bf16_kernel<<<4096, blk, 0, stream>>>(Wk, Wb, (long)D * Nkv);
  gemm_bf16_wmma<false><<<dim3(Nkv / 128, M / 128), blk, 0, stream>>>(
      Xb, Wb, nullptr, Kb, M, Nkv, D);

  // 4) V = X @ Wv
  f32_to_bf16_kernel<<<4096, blk, 0, stream>>>(Wv, Wb, (long)D * Nkv);
  gemm_bf16_wmma<false><<<dim3(Nkv / 128, M / 128), blk, 0, stream>>>(
      Xb, Wb, nullptr, Vb, M, Nkv, D);

  // 5) RoPE on Q and K (in place)
  {
    long tq = (long)M * H * 64;
    rope_kernel<<<(unsigned)((tq + 255) / 256), blk, 0, stream>>>(Qb, cosT, sinT, L, H, tq);
    long tk = (long)M * KVH * 64;
    rope_kernel<<<(unsigned)((tk + 255) / 256), blk, 0, stream>>>(Kb, cosT, sinT, L, KVH, tk);
  }

  // 6) causal flash attention -> Ab (bf16)
  flash_attn_kernel<<<dim3(L / 128, H, B), blk, 0, stream>>>(Qb, Kb, Vb, Ab, L);

  // 7) out = Ab @ Wo (fp32 output)
  f32_to_bf16_kernel<<<4096, blk, 0, stream>>>(Wo, Wb, (long)Nq * D);
  gemm_bf16_wmma<true><<<dim3(D / 128, M / 128), blk, 0, stream>>>(
      Ab, Wb, out, nullptr, M, D, Nq);
}
